// SwinFeatureEnhancer_90013924590144
// MI455X (gfx1250) — matrix operand order repaired and checked by reading, not compiled, not run on hardware
//
#include <hip/hip_runtime.h>
#include <cmath>

// ---------------------------------------------------------------------------
// Types for CDNA5 WMMA (wave32): v_wmma_f32_16x16x32_bf16
// ---------------------------------------------------------------------------
typedef __attribute__((ext_vector_type(16))) __bf16 v16bf;
typedef __attribute__((ext_vector_type(8)))  float  v8f;
typedef __attribute__((ext_vector_type(4)))  int    i4;   // POD 16-byte chunk

union FragBF {       // 16 bf16 = one A or B fragment per lane
  v16bf v;
  i4    q[2];
};

__device__ __forceinline__ __bf16 f2bf(float f) {
  union { float f; unsigned u; } a; a.f = f;
  unsigned u = a.u;
  unsigned r = (u + 0x7FFFu + ((u >> 16) & 1u)) >> 16;   // round-to-nearest-even
  union { unsigned short s; __bf16 b; } o; o.s = (unsigned short)r;
  return o.b;
}

// Problem constants (B=8, C=256, H=W=128, ws=8, heads=8, hd=32)
static constexpr long M_TOK = 131072;   // nW * 64 = 2048 * 64
static constexpr int  C_DIM = 256;

// ---------------------------------------------------------------------------
// Weight convert: f32 [out,in] -> bf16 [N][K] (layout already B-fragment ready)
// ---------------------------------------------------------------------------
__global__ __launch_bounds__(256)
void cvt_f32_bf16(const float* __restrict__ in, __bf16* __restrict__ out, long n) {
  long i = (long)blockIdx.x * 256 + threadIdx.x;
  long stride = (long)gridDim.x * 256;
  for (; i < n; i += stride) out[i] = f2bf(in[i]);
}

// ---------------------------------------------------------------------------
// LayerNorm over C=256, one token per wave (8 waves / block).
// GATHER=1: read x in [B,C,H,W] with window permutation folded in.
// GATHER=0: read f32 rows [row, 256].
// ---------------------------------------------------------------------------
template<int GATHER>
__global__ __launch_bounds__(256)
void ln_kernel(const float* __restrict__ in, const float* __restrict__ g,
               const float* __restrict__ beta, __bf16* __restrict__ out) {
  const int lane = threadIdx.x & 31;
  const int wave = threadIdx.x >> 5;
  const long row = (long)blockIdx.x * 8 + wave;

  const float* base;
  long cstride;
  if (GATHER) {
    int n  = (int)(row & 63); long w = row >> 6;
    int ix = n & 7, iy = n >> 3;
    int wx = (int)(w & 15), wy = (int)((w >> 4) & 15), bb = (int)(w >> 8);
    int y = wy * 8 + iy, xx = wx * 8 + ix;
    base = in + (((long)bb * C_DIM) * 128 + y) * 128 + xx;
    cstride = 128 * 128;          // channel stride in [B,C,H,W]
  } else {
    base = in + row * C_DIM;
    cstride = 1;
  }

  float vals[8];
  float s = 0.f;
#pragma unroll
  for (int i = 0; i < 8; ++i) {
    int c = lane + i * 32;
    vals[i] = base[(long)c * cstride];
    s += vals[i];
  }
#pragma unroll
  for (int m = 16; m > 0; m >>= 1) s += __shfl_xor(s, m, 32);
  const float mean = s * (1.0f / 256.0f);

  float vs = 0.f;
#pragma unroll
  for (int i = 0; i < 8; ++i) { float d = vals[i] - mean; vs += d * d; }
#pragma unroll
  for (int m = 16; m > 0; m >>= 1) vs += __shfl_xor(vs, m, 32);
  const float rstd = rsqrtf(vs * (1.0f / 256.0f) + 1e-5f);

#pragma unroll
  for (int i = 0; i < 8; ++i) {
    int c = lane + i * 32;
    out[row * C_DIM + c] = f2bf((vals[i] - mean) * rstd * g[c] + beta[c]);
  }
}

// ---------------------------------------------------------------------------
// Tiled WMMA GEMM with CDNA5 async LDS copies (ASYNCcnt) + double buffering.
// out[M,N] = A[M,K](bf16) * W[N,K]^T(bf16) + epilogue
// Block tile 128x64, K-step 32. 8 waves: 4(M) x 2(N), each wave 32x32 (4 WMMA).
// Per K-step each wave issues 3 global_load_async_to_lds_b128 (A:2, B:1);
// in-order ASYNCcnt completion => s_wait_asynccnt 3 retires the previous tile.
// MODE 0: +bias, store bf16
// MODE 1: +bias, exact GELU, store bf16
// MODE 3: +bias + residual gathered from x[B,C,H,W], store f32 rows
// MODE 4: +bias + residual f32 rows, scatter f32 to [B,C,H,W]
// ---------------------------------------------------------------------------
#define BM 128
#define BN 64
#define BK 32
#define LDT (BK + 8)   // padded LDS row stride (halves); 80B rows, 16B aligned

template<int MODE>
__global__ __launch_bounds__(256)
void gemm_bf16(const __bf16* __restrict__ A, const __bf16* __restrict__ Bw,
               const float* __restrict__ bias, void* __restrict__ out,
               const float* __restrict__ resid, const float* __restrict__ xsrc,
               int N, int K) {
  __shared__ __align__(16) __bf16 As[2][BM * LDT];
  __shared__ __align__(16) __bf16 Bs[2][BN * LDT];

  const int tid  = threadIdx.x;
  const int lane = tid & 31;
  const int wave = tid >> 5;
  const int wm   = wave & 3;   // wave row (M)
  const int wn   = wave >> 2;  // wave col (N)
  const long mBlock = (long)blockIdx.y * BM;
  const long nBlock = (long)blockIdx.x * BN;

  // per-thread tile-copy assignment (per-lane addresses, wave-wide async ops)
  const int r_a = tid >> 1;           // A: 128 rows, 2 threads/row
  const int c_a = (tid & 1) * 16;     // 16 halves = 32B per thread (2 x b128)
  const int r_b = tid >> 2;           // B: 64 rows, 4 threads/row
  const int c_b = (tid & 3) * 8;      // 8 halves = 16B per thread (1 x b128)

  const __bf16* gA = A + (mBlock + r_a) * (long)K + c_a;
  const __bf16* gB = Bw + (nBlock + r_b) * (long)K + c_b;

  auto issue_tile = [&](int k0, int buf) {
    unsigned la = (unsigned)(unsigned long long)(&As[buf][r_a * LDT + c_a]);
    const __bf16* pa = gA + k0;
    asm volatile("global_load_async_to_lds_b128 %0, %1, off"
                 :: "v"(la), "v"(pa) : "memory");
    asm volatile("global_load_async_to_lds_b128 %0, %1, off"
                 :: "v"(la + 16u), "v"(pa + 8) : "memory");
    unsigned lb = (unsigned)(unsigned long long)(&Bs[buf][r_b * LDT + c_b]);
    const __bf16* pb = gB + k0;
    asm volatile("global_load_async_to_lds_b128 %0, %1, off"
                 :: "v"(lb), "v"(pb) : "memory");
  };

  v8f acc[2][2] = {};

  const int arow  = lane & 15;
  const int aksel = (lane >> 4) * 8;    // A frag: lanes16-31 take K 8..15 / 24..31
  const int bksel = (lane >> 4) * 8;    // B frag: lanes16-31 take K 8..15 / 24..31

  issue_tile(0, 0);
  int buf = 0;
  for (int k0 = 0; k0 < K; k0 += BK) {
    const bool more = (k0 + BK) < K;
    if (more) {
      issue_tile(k0 + BK, buf ^ 1);
      asm volatile("s_wait_asynccnt 3" ::: "memory");  // tile(k0) resident
    } else {
      asm volatile("s_wait_asynccnt 0" ::: "memory");
    }
    __syncthreads();   // all waves' async copies for this tile visible

    FragBF a[2], b[2];
#pragma unroll
    for (int i = 0; i < 2; ++i) {
      const __bf16* p = &As[buf][(wm * 32 + i * 16 + arow) * LDT + aksel];
      a[i].q[0] = *(const i4*)(p);
      a[i].q[1] = *(const i4*)(p + 16);
    }
#pragma unroll
    for (int j = 0; j < 2; ++j) {
      const __bf16* p = &Bs[buf][(wn * 32 + j * 16 + arow) * LDT + bksel];
      b[j].q[0] = *(const i4*)(p);
      b[j].q[1] = *(const i4*)(p + 16);
    }
#pragma unroll
    for (int i = 0; i < 2; ++i)
#pragma unroll
      for (int j = 0; j < 2; ++j)
        acc[i][j] = __builtin_amdgcn_wmma_f32_16x16x32_bf16(
            false, a[i].v, false, b[j].v, (short)0, acc[i][j], false, false);
    __syncthreads();   // protect buf from tile(k0+2BK) async writes
    buf ^= 1;
  }

  // Epilogue: C/D layout: VGPR r -> row r (lanes 0-15) / r+8 (lanes 16-31)
  const long mW = mBlock + wm * 32;
  const int  nWv = (int)nBlock + wn * 32;
  const int  rofs = (lane >> 4) << 3;
#pragma unroll
  for (int i = 0; i < 2; ++i) {
#pragma unroll
    for (int j = 0; j < 2; ++j) {
#pragma unroll
      for (int r = 0; r < 8; ++r) {
        long row = mW + i * 16 + r + rofs;
        int  col = nWv + j * 16 + (lane & 15);
        float v = acc[i][j][r] + bias[col];
        if (MODE == 0) {
          ((__bf16*)out)[row * N + col] = f2bf(v);
        } else if (MODE == 1) {
          v = 0.5f * v * (1.0f + erff(v * 0.70710678118f));
          ((__bf16*)out)[row * N + col] = f2bf(v);
        } else {
          int n  = (int)(row & 63); long w = row >> 6;
          int ix = n & 7, iy = n >> 3;
          int wx = (int)(w & 15), wy = (int)((w >> 4) & 15), bb = (int)(w >> 8);
          int y = wy * 8 + iy, xx = wx * 8 + ix;
          long bchw = (((long)bb * C_DIM + col) * 128 + y) * 128 + xx;
          if (MODE == 3) {                 // residual = transposed x, store rows
            v += xsrc[bchw];
            ((float*)out)[row * N + col] = v;
          } else {                          // MODE 4: residual rows, scatter BCHW
            v += resid[row * N + col];
            ((float*)out)[bchw] = v;
          }
        }
      }
    }
  }
}

// ---------------------------------------------------------------------------
// Windowed attention. Block = 256 threads = 8 waves = 4 heads x 2 row-halves.
// grid = nW * 2 (head-groups of 4). qkv rows: [tok, 768] = [q(256)|k(256)|v(256)].
// Q,K fragments load straight from global (layouts already match WMMA);
// P (softmax) and V^T staged in LDS for the second WMMA.
// ---------------------------------------------------------------------------
__global__ __launch_bounds__(256)
void attn_kernel(const __bf16* __restrict__ qkv, const float* __restrict__ rpb,
                 const int* __restrict__ rel_idx, __bf16* __restrict__ outp) {
  const int w    = blockIdx.x >> 1;
  const int hgrp = blockIdx.x & 1;
  const int wave = threadIdx.x >> 5;
  const int lane = threadIdx.x & 31;
  const int hl   = wave >> 1;          // head-in-group 0..3
  const int half = wave & 1;           // row half 0/1 (rows half*32..+31)
  const int h    = hgrp * 4 + hl;

  __shared__ __align__(16) __bf16 Plds[4][64][72];  // probs, [head][n][m]
  __shared__ __align__(16) __bf16 Vt[4][32][72];    // V^T,  [head][d][m]

  const long tokBase = (long)w * 64;

  // --- stage V^T: each of the 2 waves of this head loads 32 token rows ---
  {
    int m = half * 32 + lane;   // token 0..63
    const __bf16* vr = qkv + (tokBase + m) * 768 + 512 + h * 32;
    union { i4 q[4]; __bf16 e[32]; } vbuf;
#pragma unroll
    for (int c = 0; c < 4; ++c) vbuf.q[c] = *(const i4*)(vr + c * 8);
#pragma unroll
    for (int d = 0; d < 32; ++d) Vt[hl][d][m] = vbuf.e[d];
  }
  __syncthreads();

  const int arow  = lane & 15;
  const int aksel = (lane >> 4) * 8;
  const int bksel = (lane >> 4) * 8;
  const int rofs  = (lane >> 4) << 3;

  // --- S = Q K^T : A frags = Q rows, B frags = K rows (natural layout) ---
  const __bf16* qbase = qkv + tokBase * 768 + h * 32;
  const __bf16* kbase = qkv + tokBase * 768 + 256 + h * 32;

  FragBF qf[2];
#pragma unroll
  for (int i = 0; i < 2; ++i) {
    const __bf16* p = qbase + (long)(half * 32 + i * 16 + arow) * 768 + aksel;
    qf[i].q[0] = *(const i4*)(p);
    qf[i].q[1] = *(const i4*)(p + 16);
  }

  v8f s[2][4] = {};
#pragma unroll
  for (int j = 0; j < 4; ++j) {
    FragBF kf;
    const __bf16* p = kbase + (long)(j * 16 + arow) * 768 + bksel;
    kf.q[0] = *(const i4*)(p);
    kf.q[1] = *(const i4*)(p + 16);
#pragma unroll
    for (int i = 0; i < 2; ++i)
      s[i][j] = __builtin_amdgcn_wmma_f32_16x16x32_bf16(
          false, qf[i].v, false, kf.v, (short)0, s[i][j], false, false);
  }

  // --- bias + softmax (rows live across lanes 0-15 / 16-31 per VGPR) ---
  const float scale = 0.17677669529663687f;  // 1/sqrt(32)
#pragma unroll
  for (int i = 0; i < 2; ++i) {
#pragma unroll
    for (int r = 0; r < 8; ++r) {
      int n = half * 32 + i * 16 + r + rofs;
      float mx = -3.0e38f;
#pragma unroll
      for (int j = 0; j < 4; ++j) {
        int m = j * 16 + (lane & 15);
        float t = s[i][j][r] * scale + rpb[rel_idx[n * 64 + m] * 8 + h];
        s[i][j][r] = t;
        mx = fmaxf(mx, t);
      }
#pragma unroll
      for (int msk = 1; msk < 16; msk <<= 1) mx = fmaxf(mx, __shfl_xor(mx, msk, 32));
      float sum = 0.f;
#pragma unroll
      for (int j = 0; j < 4; ++j) {
        float e = __expf(s[i][j][r] - mx);
        s[i][j][r] = e;
        sum += e;
      }
#pragma unroll
      for (int msk = 1; msk < 16; msk <<= 1) sum += __shfl_xor(sum, msk, 32);
      float inv = 1.0f / sum;
#pragma unroll
      for (int j = 0; j < 4; ++j) {
        int m = j * 16 + (lane & 15);
        Plds[hl][n][m] = f2bf(s[i][j][r] * inv);
      }
    }
  }
  // P rows of this wave are produced & consumed by this wave; DScnt orders it.

  // --- O = P V : A frags from Plds, B frags from Vt; K=64 in 2 steps ---
  v8f o[2][2] = {};
#pragma unroll
  for (int kk = 0; kk < 2; ++kk) {
    FragBF pa[2];
#pragma unroll
    for (int i = 0; i < 2; ++i) {
      const __bf16* p = &Plds[hl][half * 32 + i * 16 + arow][kk * 32 + aksel];
      pa[i].q[0] = *(const i4*)(p);
      pa[i].q[1] = *(const i4*)(p + 16);
    }
#pragma unroll
    for (int j = 0; j < 2; ++j) {
      FragBF vb;
      const __bf16* p = &Vt[hl][j * 16 + arow][kk * 32 + bksel];
      vb.q[0] = *(const i4*)(p);
      vb.q[1] = *(const i4*)(p + 16);
#pragma unroll
      for (int i = 0; i < 2; ++i)
        o[i][j] = __builtin_amdgcn_wmma_f32_16x16x32_bf16(
            false, pa[i].v, false, vb.v, (short)0, o[i][j], false, false);
    }
  }

  // --- store attn_out bf16 [tok, head*32 + d] ---
  __bf16* ob = outp + tokBase * C_DIM + h * 32;
#pragma unroll
  for (int i = 0; i < 2; ++i)
#pragma unroll
    for (int j = 0; j < 2; ++j)
#pragma unroll
      for (int r = 0; r < 8; ++r) {
        int n = half * 32 + i * 16 + r + rofs;
        int d = j * 16 + (lane & 15);
        ob[(long)n * C_DIM + d] = f2bf(o[i][j][r]);
      }
}

// ---------------------------------------------------------------------------
// Host orchestration
// ---------------------------------------------------------------------------
extern "C" void kernel_launch(void* const* d_in, const int* in_sizes, int n_in,
                              void* d_out, int out_size, void* d_ws, size_t ws_size,
                              hipStream_t stream) {
  const float* x      = (const float*)d_in[0];
  const float* n1g    = (const float*)d_in[1];
  const float* n1b    = (const float*)d_in[2];
  const float* qkv_w  = (const float*)d_in[3];
  const float* qkv_b  = (const float*)d_in[4];
  const float* proj_w = (const float*)d_in[5];
  const float* proj_b = (const float*)d_in[6];
  const float* rpb    = (const float*)d_in[7];
  const float* n2g    = (const float*)d_in[8];
  const float* n2b    = (const float*)d_in[9];
  const float* fc1_w  = (const float*)d_in[10];
  const float* fc1_b  = (const float*)d_in[11];
  const float* fc2_w  = (const float*)d_in[12];
  const float* fc2_b  = (const float*)d_in[13];
  const int*   relix  = (const int*)d_in[14];

  char* ws = (char*)d_ws;
  size_t off = 0;
  auto alloc = [&](size_t bytes) -> void* {
    void* p = ws + off;
    off = (off + bytes + 255) & ~(size_t)255;
    return p;
  };

  // Region A: LN1 output, later reused for LN2 output (bf16 rows [M,256])
  __bf16* hln    = (__bf16*)alloc(M_TOK * 256 * 2);
  // Region B: QKV output [M,768], later reused for FC1 output [M,1024]
  __bf16* bigbuf = (__bf16*)alloc(M_TOK * 1024 * 2);
  // Region C: attention output bf16 [M,256]
  __bf16* attout = (__bf16*)alloc(M_TOK * 256 * 2);
  // Region D: post-attention residual f32 rows [M,256]
  float*  hres   = (float*)alloc(M_TOK * 256 * 4);
  // bf16 weights (layout [out,in] == WMMA B-fragment [N][K], no transpose)
  __bf16* wq  = (__bf16*)alloc(768 * 256 * 2);
  __bf16* wp  = (__bf16*)alloc(256 * 256 * 2);
  __bf16* wf1 = (__bf16*)alloc(1024 * 256 * 2);
  __bf16* wf2 = (__bf16*)alloc(256 * 1024 * 2);

  // 0) weight converts
  cvt_f32_bf16<<<768, 256, 0, stream>>>(qkv_w, wq, 768L * 256);
  cvt_f32_bf16<<<256, 256, 0, stream>>>(proj_w, wp, 256L * 256);
  cvt_f32_bf16<<<1024, 256, 0, stream>>>(fc1_w, wf1, 1024L * 256);
  cvt_f32_bf16<<<1024, 256, 0, stream>>>(fc2_w, wf2, 256L * 1024);

  // 1) LN1 with window-partition gather from [B,C,H,W]
  ln_kernel<1><<<16384, 256, 0, stream>>>(x, n1g, n1b, hln);

  // 2) QKV GEMM: [M,256] x [256,768]
  gemm_bf16<0><<<dim3(768 / BN, M_TOK / BM), 256, 0, stream>>>(
      hln, wq, qkv_b, bigbuf, nullptr, nullptr, 768, 256);

  // 3) windowed attention (2048 windows x 2 head-groups)
  attn_kernel<<<4096, 256, 0, stream>>>(bigbuf, rpb, relix, attout);

  // 4) proj GEMM + residual (gathered transposed x) -> f32 rows
  gemm_bf16<3><<<dim3(256 / BN, M_TOK / BM), 256, 0, stream>>>(
      attout, wp, proj_b, hres, nullptr, x, 256, 256);

  // 5) LN2 on f32 rows -> bf16 rows (reuse region A)
  ln_kernel<0><<<16384, 256, 0, stream>>>(hres, n2g, n2b, hln);

  // 6) FC1 + exact GELU (reuse region B)
  gemm_bf16<1><<<dim3(1024 / BN, M_TOK / BM), 256, 0, stream>>>(
      hln, wf1, fc1_b, bigbuf, nullptr, nullptr, 1024, 256);

  // 7) FC2 + residual + scatter back to [B,C,H,W]
  gemm_bf16<4><<<dim3(256 / BN, M_TOK / BM), 256, 0, stream>>>(
      bigbuf, wf2, fc2_b, d_out, hres, nullptr, 256, 1024);
}
